// HighPerformanceSparseSimilarity_80135499809324
// MI455X (gfx1250) — compile-verified, matrix operand order checked by reading, not run on hardware
//
#include <hip/hip_runtime.h>
#include <stdint.h>

// ---------------------------------------------------------------------------
// Problem constants (B=1, Nx=Ny=16384, C=256, top-K=15, tau=0.2)
// ---------------------------------------------------------------------------
#define NROWS 16384
#define NCOLS 16384
#define CDIM  256
#define TOPK  15
#define INV_TAU 5.0f
#define NT    (NCOLS / 16)       // 1024 column tiles
#define WAVES 8                  // waves per workgroup (256 threads)
#define ROWS_PER_WG (16 * WAVES) // 128 rows per workgroup

typedef __attribute__((ext_vector_type(16))) __bf16    v16bf;
typedef __attribute__((ext_vector_type(8)))  float     v8f;
typedef __attribute__((ext_vector_type(4)))  uint32_t  v4u;

// GCC-style vector to match the async builtin's parameter type exactly
typedef int v4i_g __attribute__((vector_size(16)));

union Frag {
    v16bf bf;
    v4u   u[2];   // two 16-byte halves (8 bf16 each)
};

// --- CDNA5 async global->LDS path (ASYNCcnt), with compile-safe fallback ----
#if defined(__has_builtin)
#if __has_builtin(__builtin_amdgcn_global_load_async_to_lds_b128)
#define HAVE_ASYNC_LDS 1
#endif
#endif

#ifdef HAVE_ASYNC_LDS
// signature (from clang diagnostic): (AS1 int4*, AS3 int4*, imm, imm)
#define ASYNC_B128(gptr, lptr, off)                                          \
    __builtin_amdgcn_global_load_async_to_lds_b128(                          \
        (__attribute__((address_space(1))) v4i_g*)(gptr),                    \
        (__attribute__((address_space(3))) v4i_g*)(lptr), (off), 0)
#if __has_builtin(__builtin_amdgcn_s_wait_asynccnt)
#define WAIT_ASYNC() __builtin_amdgcn_s_wait_asynccnt(0)
#else
#define WAIT_ASYNC() asm volatile("s_wait_asynccnt 0x0" ::: "memory")
#endif
#endif

// round-to-nearest-even fp32 -> bf16
static __device__ __forceinline__ uint32_t f2bf(float f) {
    uint32_t u = __float_as_uint(f);
    uint32_t r = u + 0x7FFFu + ((u >> 16) & 1u);
    return r >> 16;
}

// ---------------------------------------------------------------------------
// Kernel 1: L2-normalize rows (256 fp32) -> packed bf16 rows in workspace.
// One wave32 per row: 8 elements / lane, shuffle-xor reduction.
// ---------------------------------------------------------------------------
__global__ __launch_bounds__(256) void k_normalize(const float* __restrict__ in,
                                                   uint16_t* __restrict__ out,
                                                   int rows) {
    int wave = (int)((blockIdx.x * blockDim.x + threadIdx.x) >> 5);
    int lane = (int)(threadIdx.x & 31);
    if (wave >= rows) return;

    const float4* p = reinterpret_cast<const float4*>(in + (size_t)wave * CDIM + lane * 8);
    float4 a = p[0];
    float4 b = p[1];

    float s = a.x * a.x;
    s = fmaf(a.y, a.y, s);
    s = fmaf(a.z, a.z, s);
    s = fmaf(a.w, a.w, s);
    s = fmaf(b.x, b.x, s);
    s = fmaf(b.y, b.y, s);
    s = fmaf(b.z, b.z, s);
    s = fmaf(b.w, b.w, s);

    #pragma unroll
    for (int off = 16; off > 0; off >>= 1)
        s += __shfl_xor(s, off, 32);

    float inv = 1.0f / fmaxf(sqrtf(s), 1e-12f);

    v4u w;
    w.x = f2bf(a.x * inv) | (f2bf(a.y * inv) << 16);
    w.y = f2bf(a.z * inv) | (f2bf(a.w * inv) << 16);
    w.z = f2bf(b.x * inv) | (f2bf(b.y * inv) << 16);
    w.w = f2bf(b.z * inv) | (f2bf(b.w * inv) << 16);

    *reinterpret_cast<v4u*>(out + (size_t)wave * CDIM + lane * 8) = w;
}

// ---------------------------------------------------------------------------
// Kernel 2: WMMA similarity GEMM + streaming top-15 + softmax + COO emit.
//
//  - 8 waves / block; each wave owns a 16-row M tile (A frags pinned in VGPRs).
//  - All 8 waves SHARE each 16-column B chunk via a double-buffered 8 KB LDS
//    stage (one barrier per tile): 128 WGs x 8 MB = 1 GB of L2 traffic
//    instead of 8 GB with per-wave fetching.
//  - Next chunk is brought in with GLOBAL_LOAD_ASYNC_TO_LDS_B128 (ASYNCcnt):
//    no VGPR round-trip, the DMA overlaps the WMMA tile.
//  - Two interleaved WMMA accumulation chains (acc0/acc1) so independent
//    WMMAs sit between dependent ones in the matrix pipe.
//  - Per-wave transpose patch is wave-private LDS: DS ops from one wave are
//    in-order, so no barrier is needed around it (17-float pitch avoids
//    bank conflicts for the 16 row-scanners).
// ---------------------------------------------------------------------------
__global__ __launch_bounds__(32 * WAVES, 1)
void k_simtopk(const uint16_t* __restrict__ Xn,
               const uint16_t* __restrict__ Yn,
               float* __restrict__ out) {
    __shared__ __align__(16) uint16_t bstage[2][16 * CDIM];   // 2 x 8 KB B chunks
    __shared__ float patch[WAVES][16 * 17];                   // per-wave C tile, padded

    const int tid     = (int)threadIdx.x;
    const int waveId  = tid >> 5;
    const int lane    = tid & 31;
    const int half    = lane >> 4;          // 0: lanes 0-15, 1: lanes 16-31
    const int l16     = lane & 15;
    const int rowBase = (int)blockIdx.x * ROWS_PER_WG + waveId * 16;
    const int myRow   = rowBase + l16;

    // ---- pin all A fragments in VGPRs (wave32 16-bit A layout: row = lane%16,
    //      K halves {0..7,16..23} for lanes<16, {8..15,24..31} for lanes>=16)
    Frag afr[8];
    {
        const uint16_t* arow = Xn + (size_t)myRow * CDIM;
        #pragma unroll
        for (int s = 0; s < 8; ++s) {
            int k0 = s * 32 + half * 8;
            afr[s].u[0] = *reinterpret_cast<const v4u*>(arow + k0);
            afr[s].u[1] = *reinterpret_cast<const v4u*>(arow + k0 + 16);
        }
    }

    // ---- register-resident running top-K (lanes 0-15 each own one row)
    float tv[TOPK];
    int   ti[TOPK];
    #pragma unroll
    for (int q = 0; q < TOPK; ++q) { tv[q] = -1e30f; ti[q] = 0; }
    float tmin = -1e30f;
    int   tminpos = 0;

    float* lt = patch[waveId];

    // ---- prologue: stage column chunk 0 into bstage[0] (32 B per thread)
#ifdef HAVE_ASYNC_LDS
    {
        const uint16_t* gsrc = Yn + (size_t)tid * 16;
        uint16_t*       ldst = &bstage[0][0] + (size_t)tid * 16;
        ASYNC_B128(gsrc, ldst, 0);
        ASYNC_B128(gsrc, ldst, 16);
        WAIT_ASYNC();
    }
#else
    {
        const v4u* g = reinterpret_cast<const v4u*>(Yn);
        v4u q0 = g[tid * 2];
        v4u q1 = g[tid * 2 + 1];
        v4u* sb = reinterpret_cast<v4u*>(&bstage[0][0]);
        sb[tid * 2]     = q0;
        sb[tid * 2 + 1] = q1;
    }
#endif
    __syncthreads();

    for (int jt = 0; jt < NT; ++jt) {
        const int cur = jt & 1;
        const int nxt = cur ^ 1;
        const int colBase = jt * 16;

        // -- bring in next chunk; whole tile of compute hides the latency
        const int jn = (jt + 1 < NT) ? (jt + 1) : jt;
        const int jp = (jt + 2 < NT) ? (jt + 2) : jt;
#ifdef HAVE_ASYNC_LDS
        {
            const uint16_t* gsrc = Yn + (size_t)jn * 16 * CDIM + (size_t)tid * 16;
            uint16_t*       ldst = &bstage[nxt][0] + (size_t)tid * 16;
            ASYNC_B128(gsrc, ldst, 0);     // async DMA: global -> LDS, ASYNCcnt
            ASYNC_B128(gsrc, ldst, 16);
        }
#else
        const v4u* gn = reinterpret_cast<const v4u*>(Yn + (size_t)jn * 16 * CDIM);
        v4u nq0 = gn[tid * 2];
        v4u nq1 = gn[tid * 2 + 1];
#endif
        __builtin_prefetch(Yn + (size_t)jp * 16 * CDIM + (size_t)tid * 16, 0, 1);

        // -- preload B fragments from LDS (column = lane%16)
        const uint16_t* bb = &bstage[cur][0] + (size_t)l16 * CDIM;
        Frag bfr[8];
        #pragma unroll
        for (int s = 0; s < 8; ++s) {
            int k0 = s * 32 + half * 8;
            bfr[s].u[0] = *reinterpret_cast<const v4u*>(bb + k0);
            bfr[s].u[1] = *reinterpret_cast<const v4u*>(bb + k0 + 16);
        }

        // -- two interleaved, independent WMMA chains over K = 256
        v8f acc0 = {};
        v8f acc1 = {};
        #pragma unroll
        for (int s = 0; s < 4; ++s) {
            acc0 = __builtin_amdgcn_wmma_f32_16x16x32_bf16(
                false, afr[2 * s].bf, false, bfr[2 * s].bf,
                (short)0, acc0, false, false);
            acc1 = __builtin_amdgcn_wmma_f32_16x16x32_bf16(
                false, afr[2 * s + 1].bf, false, bfr[2 * s + 1].bf,
                (short)0, acc1, false, false);
        }

        // -- transpose C tile through wave-private LDS patch
        //    (C layout: lane = col, VGPR v -> M = v + 8*half)
        #pragma unroll
        for (int v = 0; v < 8; ++v)
            lt[(v + 8 * half) * 17 + l16] = (acc0[v] + acc1[v]) * INV_TAU;

        // -- per-row top-K update; wave-internal DS in-order => no barrier
        if (lane < 16) {
            #pragma unroll 4
            for (int c = 0; c < 16; ++c) {
                float val = lt[lane * 17 + c];
                if (val > tmin) {
                    int col = colBase + c;
                    #pragma unroll
                    for (int q = 0; q < TOPK; ++q)
                        if (q == tminpos) { tv[q] = val; ti[q] = col; }
                    tmin = tv[0]; tminpos = 0;
                    #pragma unroll
                    for (int q = 1; q < TOPK; ++q)
                        if (tv[q] < tmin) { tmin = tv[q]; tminpos = q; }
                }
            }
        }

        // -- make next chunk visible, then one barrier
#ifdef HAVE_ASYNC_LDS
        WAIT_ASYNC();                       // ASYNCcnt -> 0: LDS writes landed
#else
        {
            v4u* sb = reinterpret_cast<v4u*>(&bstage[nxt][0]);
            sb[tid * 2]     = nq0;
            sb[tid * 2 + 1] = nq1;
        }
#endif
        __syncthreads();
    }

    // ---- finalize: sort descending (matches lax.top_k order), softmax, emit COO
    if (lane < 16) {
        #pragma unroll
        for (int a = 0; a < TOPK - 1; ++a) {
            #pragma unroll
            for (int b = a + 1; b < TOPK; ++b) {
                if (tv[b] > tv[a]) {
                    float fv = tv[a]; tv[a] = tv[b]; tv[b] = fv;
                    int   fi = ti[a]; ti[a] = ti[b]; ti[b] = fi;
                }
            }
        }

        float m = tv[0];
        float e[TOPK];
        float sum = 0.0f;
        #pragma unroll
        for (int q = 0; q < TOPK; ++q) { e[q] = __expf(tv[q] - m); sum += e[q]; }
        float invs = 1.0f / sum;

        const size_t NK   = (size_t)NROWS * TOPK;
        const size_t base = (size_t)myRow * TOPK;
        #pragma unroll
        for (int q = 0; q < TOPK; ++q) {
            out[0 * NK + base + q] = 0.0f;              // b index (B == 1)
            out[1 * NK + base + q] = (float)myRow;      // row index
            out[2 * NK + base + q] = (float)ti[q];      // col index
            out[3 * NK + base + q] = e[q] * invs;       // softmax value
        }
    }
}

// ---------------------------------------------------------------------------
// Host-side launch glue (all on `stream`, graph-capture safe)
// ---------------------------------------------------------------------------
extern "C" void kernel_launch(void* const* d_in, const int* in_sizes, int n_in,
                              void* d_out, int out_size, void* d_ws, size_t ws_size,
                              hipStream_t stream) {
    (void)in_sizes; (void)n_in; (void)out_size; (void)ws_size;

    const float* feat_x = (const float*)d_in[0];
    const float* feat_y = (const float*)d_in[1];
    float* out = (float*)d_out;

    uint16_t* Xn = (uint16_t*)d_ws;                        // 16384*256 bf16 = 8 MB
    uint16_t* Yn = Xn + (size_t)NROWS * CDIM;              // next 8 MB

    // normalize: 8 rows (waves) per 256-thread block
    dim3 nblk(NROWS / 8);
    k_normalize<<<nblk, 256, 0, stream>>>(feat_x, Xn, NROWS);
    k_normalize<<<nblk, 256, 0, stream>>>(feat_y, Yn, NCOLS);

    // similarity + top-k: 8 waves/block * 16 rows/wave = 128 rows/block
    dim3 gblk(NROWS / ROWS_PER_WG);
    k_simtopk<<<gblk, 32 * WAVES, 0, stream>>>(Xn, Yn, out);
}
